// ScopeAwareT5Block_14456859919010
// MI455X (gfx1250) — matrix operand order repaired and checked by reading, not compiled, not run on hardware
//
#include <hip/hip_runtime.h>
#include <hip/hip_bf16.h>
#include <cstdint>
#include <cstddef>

// ---------------------------------------------------------------------------
// ScopeAwareT5Block for MI455X (gfx1250, wave32, WMMA).
//
// Precision strategy: all GEMMs run on v_wmma_f32_16x16x32_f16 (f16 operands,
// f32 accumulation). Softmax, norms, residuals, masks in fp32.
//
// Input flattening assumption (recursive insertion order of setup_inputs()):
//   0-8   code_sa:   Wq bq Wk bk Wv bv Wo bo ln_w
//   9-17  pragma_sa: Wq bq Wk bk Wv bv Wo bo ln_w
//   18-27 c2p:       Wq bq Wk bk Wv bv Wo bo ln_w ln_b
//   28-37 p2c:       Wq bq Wk bk Wv bv Wo bo ln_w ln_b
//   38-40 code_ff:   ln_w wi wo
//   41-43 pragma_ff: ln_w wi wo
//   44 code_input  45 pragma_input  46 code_mask(bool8) 47 pragma_mask(bool8)
//   48 scope_mask [B,1,P,C] f32      49 alpha (f32 scalar on device)
// ---------------------------------------------------------------------------

typedef __attribute__((ext_vector_type(16))) _Float16 v16h;
typedef __attribute__((ext_vector_type(8)))  _Float16 v8h;
typedef __attribute__((ext_vector_type(8)))  float    v8f;
typedef __attribute__((ext_vector_type(4)))  int      v4i;

constexpr int DMODEL = 1024;
constexpr int NHEAD  = 16;
constexpr int HDIM   = 64;
constexpr int FFDIM  = 4096;
constexpr int BATCH  = 4;
constexpr int CSEQ   = 1024;
constexpr int PSEQ   = 512;
constexpr int BC     = BATCH * CSEQ;   // 4096 rows
constexpr int BP     = BATCH * PSEQ;   // 2048 rows

__device__ __forceinline__ v8f wmma_f16(v16h a, v16h b, v8f c) {
  return __builtin_amdgcn_wmma_f32_16x16x32_f16(false, a, false, b, (short)0, c,
                                                false, false);
}

// ---------------------------------------------------------------------------
// DPP16 row reductions (16-lane rows match the WMMA C-layout N groups).
// row_ror:n = dpp_ctrl 0x120|n; combining over 8/4/2/1 fully reduces the row.
// ---------------------------------------------------------------------------
template <int N>
__device__ __forceinline__ float dpp_ror16(float v) {
  const int x = __float_as_int(v);
  return __int_as_float(
      __builtin_amdgcn_update_dpp(x, x, 0x120 | N, 0xF, 0xF, false));
}
__device__ __forceinline__ float rmax16(float v) {
  v = fmaxf(v, dpp_ror16<8>(v));
  v = fmaxf(v, dpp_ror16<4>(v));
  v = fmaxf(v, dpp_ror16<2>(v));
  v = fmaxf(v, dpp_ror16<1>(v));
  return v;
}
__device__ __forceinline__ float rsum16(float v) {
  v += dpp_ror16<8>(v);
  v += dpp_ror16<4>(v);
  v += dpp_ror16<2>(v);
  v += dpp_ror16<1>(v);
  return v;
}

// ---------------------------------------------------------------------------
// gfx1250 async global->LDS staging (ASYNCcnt tracked), with sync fallback.
// Builtin parameter types (from hipcc diagnostic): v4i in AS1 / AS3.
// ---------------------------------------------------------------------------
#if __has_builtin(__builtin_amdgcn_global_load_async_to_lds_b128)
typedef __attribute__((address_space(1))) v4i* as1_v4i;
typedef __attribute__((address_space(3))) v4i* as3_v4i;
__device__ __forceinline__ void stage16(const float* g, float* l) {
  __builtin_amdgcn_global_load_async_to_lds_b128((as1_v4i)g, (as3_v4i)l, 0, 0);
}
__device__ __forceinline__ void stage_wait() {
#if __has_builtin(__builtin_amdgcn_s_wait_asynccnt)
  __builtin_amdgcn_s_wait_asynccnt(0);
#else
  asm volatile("s_wait_asynccnt 0x0" ::: "memory");
#endif
}
#else
__device__ __forceinline__ void stage16(const float* g, float* l) {
  *(float4*)l = *(const float4*)g;
}
__device__ __forceinline__ void stage_wait() {}
#endif

// A-fragment (16x32 f16): lane holds row m = lane&15; half = lane>>4 selects
// contiguous K range {half*16..+16} (matches the B fragments).
__device__ __forceinline__ v16h load_a16(const _Float16* row, int kb, int half) {
  v8h lo = *(const v8h*)(row + kb + half * 16);
  v8h hi = *(const v8h*)(row + kb + half * 16 + 8);
  v16h a;
#pragma unroll
  for (int i = 0; i < 8; ++i) { a[i] = lo[i]; a[i + 8] = hi[i]; }
  return a;
}

// B-fragment (32x16 f16): b[i] = B[k = i + 16*half][n = lane&15]. Caller passes
// a pointer to 16 contiguous halves along K for its lane (B = W^T row-major).
__device__ __forceinline__ v16h load_b16(const _Float16* p) {
  v8h lo = *(const v8h*)(p);
  v8h hi = *(const v8h*)(p + 8);
  v16h b;
#pragma unroll
  for (int i = 0; i < 8; ++i) { b[i] = lo[i]; b[i + 8] = hi[i]; }
  return b;
}

// ---------------------------------------------------------------------------
// Generic WMMA GEMM: out[M,N] = A_f16[M,K] * W[N,K]^T (+bias)(+resid)(relu)
// 128 threads = 4 waves; block tile 64x64; wave tile 16x64; K-step 32.
// The 64x32 fp32 W tile is shared by all 4 waves, so it is staged once per
// block into LDS via the async DMA path, double-buffered so the copy of tile
// k+1 overlaps the WMMAs of tile k.
// QKV=true scatters the f16 output into [B, H, S, HDIM] layout.
// ---------------------------------------------------------------------------
template <bool BIAS, bool RESID, bool RELU, bool OUT32, bool OUT16, bool QKV>
__global__ __launch_bounds__(128) void gemm_k(
    const _Float16* __restrict__ A, const float* __restrict__ W,
    const float* __restrict__ bias, const float* resid,
    float* out32, _Float16* out16, int M, int N, int K, int S) {
  const int tid  = threadIdx.x;
  const int lane = tid & 31;
  const int nl   = lane & 15;
  const int half = lane >> 4;
  const int wave = tid >> 5;
  const int n0   = blockIdx.x * 64;
  const int mb   = blockIdx.y * 64 + wave * 16;

  __shared__ __align__(16) float wtile[2][64 * 32];

  // stage a 64(n) x 32(k) fp32 W tile: 512 16B chunks over 128 threads
  auto stage = [&](int buf, int k0) {
#pragma unroll
    for (int j = 0; j < 4; ++j) {
      const int c = tid + j * 128;
      const int row = c >> 3, seg = c & 7;
      stage16(W + (size_t)(n0 + row) * K + k0 + seg * 4,
              &wtile[buf][row * 32 + seg * 4]);
    }
  };

  v8f acc[4];
#pragma unroll
  for (int t = 0; t < 4; ++t)
#pragma unroll
    for (int i = 0; i < 8; ++i) acc[t][i] = 0.f;

  const _Float16* arow = A + (size_t)(mb + nl) * K;
  stage(0, 0);
  int buf = 0;
  for (int k0 = 0; k0 < K; k0 += 32) {
    stage_wait();        // current buffer's DMA complete (per-wave ASYNCcnt)
    __syncthreads();     // all waves see it / done reading the other buffer
    if (k0 + 32 < K) stage(buf ^ 1, k0 + 32);

    v16h af = load_a16(arow, k0, half);
#pragma unroll
    for (int t = 0; t < 4; ++t) {
      const float* wp = &wtile[buf][(t * 16 + nl) * 32 + 16 * half];
      v16h bf;
#pragma unroll
      for (int i = 0; i < 16; ++i) bf[i] = (_Float16)wp[i];
      acc[t] = wmma_f16(af, bf, acc[t]);
    }
    buf ^= 1;
  }

#pragma unroll
  for (int t = 0; t < 4; ++t) {
    const int n = n0 + t * 16 + nl;
#pragma unroll
    for (int r = 0; r < 8; ++r) {
      const int m = mb + r + 8 * half;   // C layout row (ISA 7.12.2)
      float v = acc[t][r];
      if (BIAS)  v += bias[n];
      if (RELU)  v = fmaxf(v, 0.f);
      if (RESID) v += resid[(size_t)m * N + n];
      if (OUT32) out32[(size_t)m * N + n] = v;
      if (OUT16) {
        if (QKV) {
          const int b = m / S, s = m - b * S;
          const int h = n >> 6, hd = n & 63;
          out16[((((size_t)b * NHEAD + h) * S + s) << 6) + hd] = (_Float16)v;
        } else {
          out16[(size_t)m * N + n] = (_Float16)v;
        }
      }
    }
  }
}

// ---------------------------------------------------------------------------
// Flash self-attention: 1 wave per (b, h, 16-query tile); streams 32 keys/iter,
// online softmax with running max (key padding bias = -1e30).
// Q/K/V are f16 [B, H, S, HDIM]; ctx written f16 [B*S, DMODEL].
// ---------------------------------------------------------------------------
__global__ __launch_bounds__(32) void flash_self_k(
    const _Float16* __restrict__ Q, const _Float16* __restrict__ Kt,
    const _Float16* __restrict__ V, const unsigned char* __restrict__ kmask,
    _Float16* __restrict__ ctx, int S) {
  const int lane = threadIdx.x;
  const int nl = lane & 15, half = lane >> 4;
  const int b = blockIdx.z, h = blockIdx.y;
  const int q0 = blockIdx.x * 16;
  const size_t bh = ((size_t)b * NHEAD + h) * S;

  const _Float16* qrow = Q + (bh + q0 + nl) * HDIM;
  v16h qa0 = load_a16(qrow, 0, half);
  v16h qa1 = load_a16(qrow, 32, half);

  float mr[8], lr[8];
  v8f o[4];
#pragma unroll
  for (int r = 0; r < 8; ++r) { mr[r] = -3.0e38f; lr[r] = 0.f; }
#pragma unroll
  for (int t = 0; t < 4; ++t)
#pragma unroll
    for (int i = 0; i < 8; ++i) o[t][i] = 0.f;

  __shared__ __align__(32) _Float16 pl[16 * 32];

  for (int kt = 0; kt < S; kt += 32) {
    v8f s0, s1;
#pragma unroll
    for (int i = 0; i < 8; ++i) { s0[i] = 0.f; s1[i] = 0.f; }
    {
      const _Float16* k0p = Kt + (bh + kt + nl) * HDIM + 16 * half;
      const _Float16* k1p = Kt + (bh + kt + 16 + nl) * HDIM + 16 * half;
      s0 = wmma_f16(qa0, load_b16(k0p), s0);
      s0 = wmma_f16(qa1, load_b16(k0p + 32), s0);
      s1 = wmma_f16(qa0, load_b16(k1p), s1);
      s1 = wmma_f16(qa1, load_b16(k1p + 32), s1);
    }
    const float bias0 = kmask[(size_t)b * S + kt + nl]      ? -1.0e30f : 0.f;
    const float bias1 = kmask[(size_t)b * S + kt + 16 + nl] ? -1.0e30f : 0.f;

    float p0[8], p1[8], fsc[8];
#pragma unroll
    for (int r = 0; r < 8; ++r) {
      const float a0 = s0[r] * 0.125f + bias0;
      const float a1 = s1[r] * 0.125f + bias1;
      const float rm   = rmax16(fmaxf(a0, a1));
      const float mnew = fmaxf(mr[r], rm);
      const float f    = __expf(mr[r] - mnew);
      p0[r] = __expf(a0 - mnew);
      p1[r] = __expf(a1 - mnew);
      lr[r] = lr[r] * f + rsum16(p0[r] + p1[r]);
      mr[r] = mnew;
      fsc[r] = f;
    }
#pragma unroll
    for (int t = 0; t < 4; ++t)
#pragma unroll
      for (int r = 0; r < 8; ++r) o[t][r] *= fsc[r];

    // C-layout -> A-layout for P through LDS (single-wave workgroup: barriers
    // are S_NOP but act as compiler fences + dscnt waits).
#pragma unroll
    for (int r = 0; r < 8; ++r) {
      pl[(r + 8 * half) * 32 + nl]      = (_Float16)p0[r];
      pl[(r + 8 * half) * 32 + 16 + nl] = (_Float16)p1[r];
    }
    __syncthreads();
    v16h pa = load_a16(pl + nl * 32, 0, half);
    __syncthreads();

#pragma unroll
    for (int t = 0; t < 4; ++t) {
      v16h vf;
#pragma unroll
      for (int i = 0; i < 16; ++i)
        vf[i] = V[(bh + kt + 16 * half + i) * HDIM + t * 16 + nl];
      o[t] = wmma_f16(pa, vf, o[t]);
    }
  }

#pragma unroll
  for (int r = 0; r < 8; ++r) {
    const float inv = 1.f / lr[r];
    const int s = q0 + r + 8 * half;
#pragma unroll
    for (int t = 0; t < 4; ++t)
      ctx[((size_t)b * S + s) * DMODEL + h * HDIM + t * 16 + nl] =
          (_Float16)(o[t][r] * inv);
  }
}

// ---------------------------------------------------------------------------
// Flash cross-attention with alpha-blended scope mask + clamp(+-50).
// Clamp bounds scores, so softmax needs no running max: p = exp(s) directly.
// scope index: b*(P*C) + q*sq + k*sk  (c2p: sq=1, sk=C; p2c: sq=C, sk=1).
// ---------------------------------------------------------------------------
__global__ __launch_bounds__(32) void flash_cross_k(
    const _Float16* __restrict__ Q, const _Float16* __restrict__ Kt,
    const _Float16* __restrict__ V, const float* __restrict__ scope,
    const float* __restrict__ alphap, _Float16* __restrict__ ctx,
    int Sq, int Sk, int sq, int sk) {
  const int lane = threadIdx.x;
  const int nl = lane & 15, half = lane >> 4;
  const int b = blockIdx.z, h = blockIdx.y;
  const int q0 = blockIdx.x * 16;
  const size_t bhq = ((size_t)b * NHEAD + h) * Sq;
  const size_t bhk = ((size_t)b * NHEAD + h) * Sk;
  const float alpha = *alphap;
  const size_t mbase = (size_t)b * PSEQ * CSEQ;

  const _Float16* qrow = Q + (bhq + q0 + nl) * HDIM;
  v16h qa0 = load_a16(qrow, 0, half);
  v16h qa1 = load_a16(qrow, 32, half);

  float lr[8];
  v8f o[4];
#pragma unroll
  for (int r = 0; r < 8; ++r) lr[r] = 0.f;
#pragma unroll
  for (int t = 0; t < 4; ++t)
#pragma unroll
    for (int i = 0; i < 8; ++i) o[t][i] = 0.f;

  __shared__ __align__(32) _Float16 pl[16 * 32];

  for (int kt = 0; kt < Sk; kt += 32) {
    v8f s0, s1;
#pragma unroll
    for (int i = 0; i < 8; ++i) { s0[i] = 0.f; s1[i] = 0.f; }
    {
      const _Float16* k0p = Kt + (bhk + kt + nl) * HDIM + 16 * half;
      const _Float16* k1p = Kt + (bhk + kt + 16 + nl) * HDIM + 16 * half;
      s0 = wmma_f16(qa0, load_b16(k0p), s0);
      s0 = wmma_f16(qa1, load_b16(k0p + 32), s0);
      s1 = wmma_f16(qa0, load_b16(k1p), s1);
      s1 = wmma_f16(qa1, load_b16(k1p + 32), s1);
    }
    float p0[8], p1[8];
#pragma unroll
    for (int r = 0; r < 8; ++r) {
      const int q = q0 + r + 8 * half;
      const float m0 = scope[mbase + (size_t)q * sq + (size_t)(kt + nl) * sk];
      const float m1 = scope[mbase + (size_t)q * sq + (size_t)(kt + 16 + nl) * sk];
      float a0 = s0[r] * 0.125f;
      float a1 = s1[r] * 0.125f;
      a0 *= (1.f - alpha) + alpha * m0;
      a1 *= (1.f - alpha) + alpha * m1;
      a0 = fminf(fmaxf(a0, -50.f), 50.f);
      a1 = fminf(fmaxf(a1, -50.f), 50.f);
      p0[r] = __expf(a0);
      p1[r] = __expf(a1);
      lr[r] += rsum16(p0[r] + p1[r]);
    }
#pragma unroll
    for (int r = 0; r < 8; ++r) {
      pl[(r + 8 * half) * 32 + nl]      = (_Float16)p0[r];
      pl[(r + 8 * half) * 32 + 16 + nl] = (_Float16)p1[r];
    }
    __syncthreads();
    v16h pa = load_a16(pl + nl * 32, 0, half);
    __syncthreads();
#pragma unroll
    for (int t = 0; t < 4; ++t) {
      v16h vf;
#pragma unroll
      for (int i = 0; i < 16; ++i)
        vf[i] = V[(bhk + kt + 16 * half + i) * HDIM + t * 16 + nl];
      o[t] = wmma_f16(pa, vf, o[t]);
    }
  }

#pragma unroll
  for (int r = 0; r < 8; ++r) {
    const float inv = 1.f / lr[r];
    const int s = q0 + r + 8 * half;
#pragma unroll
    for (int t = 0; t < 4; ++t)
      ctx[((size_t)b * Sq + s) * DMODEL + h * HDIM + t * 16 + nl] =
          (_Float16)(o[t][r] * inv);
  }
}

// ---------------------------------------------------------------------------
// Row-wise norms (one block per row of D=1024), f16 outputs feed the GEMM A.
// ---------------------------------------------------------------------------
__global__ __launch_bounds__(256) void rmsnorm_k(const float* __restrict__ x,
                                                 const float* __restrict__ w,
                                                 _Float16* __restrict__ o) {
  const int row = blockIdx.x;
  const float* xr = x + (size_t)row * DMODEL;
  float ss = 0.f;
  for (int i = threadIdx.x; i < DMODEL; i += 256) { float v = xr[i]; ss += v * v; }
#pragma unroll
  for (int off = 16; off > 0; off >>= 1) ss += __shfl_down(ss, off, 32);
  __shared__ float sacc[8];
  __shared__ float sinv;
  if ((threadIdx.x & 31) == 0) sacc[threadIdx.x >> 5] = ss;
  __syncthreads();
  if (threadIdx.x == 0) {
    float t = 0.f;
#pragma unroll
    for (int i = 0; i < 8; ++i) t += sacc[i];
    sinv = rsqrtf(t * (1.f / DMODEL) + 1e-6f);
  }
  __syncthreads();
  const float inv = sinv;
  for (int i = threadIdx.x; i < DMODEL; i += 256)
    o[(size_t)row * DMODEL + i] = (_Float16)(w[i] * xr[i] * inv);
}

__global__ __launch_bounds__(256) void layernorm_k(
    const float* __restrict__ x, const float* __restrict__ w,
    const float* __restrict__ bb, float* __restrict__ o32,
    _Float16* __restrict__ o16) {
  const int row = blockIdx.x;
  const float* xr = x + (size_t)row * DMODEL;
  float s = 0.f, s2 = 0.f;
  for (int i = threadIdx.x; i < DMODEL; i += 256) {
    float v = xr[i]; s += v; s2 += v * v;
  }
#pragma unroll
  for (int off = 16; off > 0; off >>= 1) {
    s  += __shfl_down(s,  off, 32);
    s2 += __shfl_down(s2, off, 32);
  }
  __shared__ float sa[8], sb[8];
  __shared__ float smu, sinv;
  if ((threadIdx.x & 31) == 0) { sa[threadIdx.x >> 5] = s; sb[threadIdx.x >> 5] = s2; }
  __syncthreads();
  if (threadIdx.x == 0) {
    float ts = 0.f, ts2 = 0.f;
#pragma unroll
    for (int i = 0; i < 8; ++i) { ts += sa[i]; ts2 += sb[i]; }
    const float mu = ts * (1.f / DMODEL);
    const float var = ts2 * (1.f / DMODEL) - mu * mu;
    smu = mu;
    sinv = rsqrtf(var + 1e-5f);
  }
  __syncthreads();
  const float mu = smu, inv = sinv;
  for (int i = threadIdx.x; i < DMODEL; i += 256) {
    const float val = w[i] * ((xr[i] - mu) * inv) + bb[i];
    o32[(size_t)row * DMODEL + i] = val;
    o16[(size_t)row * DMODEL + i] = (_Float16)val;
  }
}

// ---------------------------------------------------------------------------
extern "C" void kernel_launch(void* const* d_in, const int* in_sizes, int n_in,
                              void* d_out, int out_size, void* d_ws,
                              size_t ws_size, hipStream_t stream) {
  (void)in_sizes; (void)n_in; (void)out_size; (void)ws_size;
  auto F = [&](int i) { return (const float*)d_in[i]; };

  // params (see flattening assumption at top)
  const float *csa_Wq = F(0),  *csa_bq = F(1),  *csa_Wk = F(2),  *csa_bk = F(3);
  const float *csa_Wv = F(4),  *csa_bv = F(5),  *csa_Wo = F(6),  *csa_bo = F(7);
  const float *csa_ln = F(8);
  const float *psa_Wq = F(9),  *psa_bq = F(10), *psa_Wk = F(11), *psa_bk = F(12);
  const float *psa_Wv = F(13), *psa_bv = F(14), *psa_Wo = F(15), *psa_bo = F(16);
  const float *psa_ln = F(17);
  const float *c2p_Wq = F(18), *c2p_bq = F(19), *c2p_Wk = F(20), *c2p_bk = F(21);
  const float *c2p_Wv = F(22), *c2p_bv = F(23), *c2p_Wo = F(24), *c2p_bo = F(25);
  const float *c2p_lnw = F(26), *c2p_lnb = F(27);
  const float *p2c_Wq = F(28), *p2c_bq = F(29), *p2c_Wk = F(30), *p2c_bk = F(31);
  const float *p2c_Wv = F(32), *p2c_bv = F(33), *p2c_Wo = F(34), *p2c_bo = F(35);
  const float *p2c_lnw = F(36), *p2c_lnb = F(37);
  const float *cff_ln = F(38), *cff_wi = F(39), *cff_wo = F(40);
  const float *pff_ln = F(41), *pff_wi = F(42), *pff_wo = F(43);
  const float *code_in = F(44), *prag_in = F(45);
  const unsigned char* code_mask = (const unsigned char*)d_in[46];
  const unsigned char* prag_mask = (const unsigned char*)d_in[47];
  const float *scope = F(48), *alpha = F(49);

  // workspace carve (all 256B aligned)
  char* wsp = (char*)d_ws;
  auto take = [&](size_t bytes) -> void* {
    void* r = (void*)wsp;
    wsp += (bytes + 255) & ~(size_t)255;
    return r;
  };
  _Float16* normA = (_Float16*)take((size_t)BC * DMODEL * 2);
  _Float16* codeH = (_Float16*)take((size_t)BC * DMODEL * 2);
  _Float16* pragH = (_Float16*)take((size_t)BP * DMODEL * 2);
  _Float16* qb    = (_Float16*)take((size_t)BC * DMODEL * 2);
  _Float16* kb    = (_Float16*)take((size_t)BC * DMODEL * 2);
  _Float16* vb    = (_Float16*)take((size_t)BC * DMODEL * 2);
  _Float16* ctxH  = (_Float16*)take((size_t)BC * DMODEL * 2);
  float*    tmp32 = (float*)take((size_t)BC * DMODEL * 4);
  _Float16* hidH  = (_Float16*)take((size_t)BC * FFDIM * 2);

  float* codeF = (float*)d_out;                       // running code [BC, D]
  float* pragF = codeF + (size_t)BC * DMODEL;         // running pragma [BP, D]

  (void)hipMemcpyAsync(codeF, code_in, (size_t)BC * DMODEL * 4,
                       hipMemcpyDeviceToDevice, stream);
  (void)hipMemcpyAsync(pragF, prag_in, (size_t)BP * DMODEL * 4,
                       hipMemcpyDeviceToDevice, stream);

  const dim3 blk(128);
  auto qkv = [&](const _Float16* A, const float* W, const float* b,
                 _Float16* out, int M, int S) {
    gemm_k<true, false, false, false, true, true>
        <<<dim3(DMODEL / 64, M / 64), blk, 0, stream>>>(
            A, W, b, nullptr, nullptr, out, M, DMODEL, DMODEL, S);
  };

  auto self_attn = [&](float* xF, _Float16* xH, const float* Wq, const float* bq,
                       const float* Wk, const float* bk2, const float* Wv,
                       const float* bv, const float* Wo, const float* bo,
                       const float* lnw, const unsigned char* mask, int S) {
    const int M = BATCH * S;
    rmsnorm_k<<<M, 256, 0, stream>>>(xF, lnw, normA);
    qkv(normA, Wq, bq, qb, M, S);
    qkv(normA, Wk, bk2, kb, M, S);
    qkv(normA, Wv, bv, vb, M, S);
    flash_self_k<<<dim3(S / 16, NHEAD, BATCH), 32, 0, stream>>>(
        qb, kb, vb, mask, ctxH, S);
    gemm_k<true, true, false, true, true, false>
        <<<dim3(DMODEL / 64, M / 64), blk, 0, stream>>>(
            ctxH, Wo, bo, xF, xF, xH, M, DMODEL, DMODEL, S);
  };

  auto cross_attn = [&](float* qF, _Float16* qH, _Float16* kH, int Sq, int Sk,
                        int sq, int sk, const float* Wq, const float* bq,
                        const float* Wk, const float* bk2, const float* Wv,
                        const float* bv, const float* Wo, const float* bo,
                        const float* lnw, const float* lnb) {
    const int Mq = BATCH * Sq, Mk = BATCH * Sk;
    qkv(qH, Wq, bq, qb, Mq, Sq);
    qkv(kH, Wk, bk2, kb, Mk, Sk);
    qkv(kH, Wv, bv, vb, Mk, Sk);
    flash_cross_k<<<dim3(Sq / 16, NHEAD, BATCH), 32, 0, stream>>>(
        qb, kb, vb, scope, alpha, ctxH, Sq, Sk, sq, sk);
    gemm_k<true, true, false, true, false, false>
        <<<dim3(DMODEL / 64, Mq / 64), blk, 0, stream>>>(
            ctxH, Wo, bo, qF, tmp32, nullptr, Mq, DMODEL, DMODEL, Sq);
    layernorm_k<<<Mq, 256, 0, stream>>>(tmp32, lnw, lnb, qF, qH);
  };

  auto ff = [&](float* xF, const float* lnw, const float* wi, const float* wo,
                int S) {
    const int M = BATCH * S;
    rmsnorm_k<<<M, 256, 0, stream>>>(xF, lnw, normA);
    gemm_k<false, false, true, false, true, false>
        <<<dim3(FFDIM / 64, M / 64), blk, 0, stream>>>(
            normA, wi, nullptr, nullptr, nullptr, hidH, M, FFDIM, DMODEL, S);
    gemm_k<false, true, false, true, false, false>
        <<<dim3(DMODEL / 64, M / 64), blk, 0, stream>>>(
            hidH, wo, nullptr, xF, xF, nullptr, M, DMODEL, FFDIM, S);
  };

  // ---- forward pass -------------------------------------------------------
  self_attn(codeF, codeH, csa_Wq, csa_bq, csa_Wk, csa_bk, csa_Wv, csa_bv,
            csa_Wo, csa_bo, csa_ln, code_mask, CSEQ);
  self_attn(pragF, pragH, psa_Wq, psa_bq, psa_Wk, psa_bk, psa_Wv, psa_bv,
            psa_Wo, psa_bo, psa_ln, prag_mask, PSEQ);
  // code -> pragma cross: query=code (q in C), key=pragma (k in P),
  // mask = scope^T  => scope[b, k, q] : sq=1, sk=CSEQ
  cross_attn(codeF, codeH, pragH, CSEQ, PSEQ, 1, CSEQ, c2p_Wq, c2p_bq, c2p_Wk,
             c2p_bk, c2p_Wv, c2p_bv, c2p_Wo, c2p_bo, c2p_lnw, c2p_lnb);
  // pragma -> code cross: query=pragma (q in P), key=code (k in C),
  // mask = scope[b, q, k] : sq=CSEQ, sk=1
  cross_attn(pragF, pragH, codeH, PSEQ, CSEQ, CSEQ, 1, p2c_Wq, p2c_bq, p2c_Wk,
             p2c_bk, p2c_Wv, p2c_bv, p2c_Wo, p2c_bo, p2c_lnw, p2c_lnb);
  ff(codeF, cff_ln, cff_wi, cff_wo, CSEQ);
  ff(pragF, pff_ln, pff_wi, pff_wo, PSEQ);
}
